// Qwen3MoeGroupedGEMMBlock_7670811591361
// MI455X (gfx1250) — matrix operand order repaired and checked by reading, not compiled, not run on hardware
//
#include <hip/hip_runtime.h>

#define T_TOKENS 4096
#define HID      1024
#define NEXP     64
#define TOPK     8
#define INTERMED 768
#define MT       128      // rows per M-tile (grouped GEMM)
#define MAXTILES 320      // sum ceil(cnt/128) <= 32768/128 + 64

typedef __attribute__((ext_vector_type(16))) __bf16 v16bf;
typedef __attribute__((ext_vector_type(8)))  float  v8f;
typedef int v4i_vs __attribute__((vector_size(16)));   // matches builtin's "V4i"

union FragBF { v16bf v; uint4 q[2]; };

__device__ __forceinline__ unsigned short f2bf(float f) {
  unsigned u = __float_as_uint(f);
  u += 0x7fffu + ((u >> 16) & 1u);           // round-to-nearest-even
  return (unsigned short)(u >> 16);
}

__device__ __forceinline__ unsigned pk2bf(float x, float y) {
  return (unsigned)f2bf(x) | ((unsigned)f2bf(y) << 16);
}

__device__ __forceinline__ uint4 pk8bf(float4 a, float4 b) {
  uint4 o;
  o.x = pk2bf(a.x, a.y);
  o.y = pk2bf(a.z, a.w);
  o.z = pk2bf(b.x, b.y);
  o.w = pk2bf(b.z, b.w);
  return o;
}

// async global -> LDS copy of 16B (CDNA5 ASYNCcnt path)
__device__ __forceinline__ void async_copy_b128(const void* g, void* l) {
#if __has_builtin(__builtin_amdgcn_global_load_async_to_lds_b128)
  __builtin_amdgcn_global_load_async_to_lds_b128(
      (__attribute__((address_space(1))) v4i_vs*)(void*)g,
      (__attribute__((address_space(3))) v4i_vs*)l, 0, 0);
#else
  asm volatile("global_load_async_to_lds_b128 %0, %1, off"
               :
               : "v"((__attribute__((address_space(3))) unsigned int*)l),
                 "v"(g)
               : "memory");
#endif
}

__device__ __forceinline__ void wait_async0() {
#if __has_builtin(__builtin_amdgcn_s_wait_asynccnt)
  __builtin_amdgcn_s_wait_asynccnt(0);
#else
  asm volatile("s_wait_asynccnt 0x0" ::: "memory");
#endif
}

__device__ __forceinline__ v8f wmma_bf16(v16bf a, v16bf b, v8f c) {
  return __builtin_amdgcn_wmma_f32_16x16x32_bf16(false, a, false, b, (short)0, c,
                                                 false, false);
}

// ---------------- init ----------------
__global__ void k_init(int* hist, int* cursor) {
  int i = threadIdx.x;
  if (i < NEXP) { hist[i] = 0; cursor[i] = 0; }
}

// ---------------- one-shot fp32 -> bf16 stream convert ----------------
__global__ __launch_bounds__(256) void k_cvt(const float* __restrict__ src,
                                             unsigned short* __restrict__ dst,
                                             long n4) {   // count of float4 groups
  long i = (long)blockIdx.x * blockDim.x + threadIdx.x;
  const long stride = (long)gridDim.x * blockDim.x;
  for (; i < n4; i += stride) {
    float4 f = ((const float4*)src)[i];
    uint2 o;
    o.x = pk2bf(f.x, f.y);
    o.y = pk2bf(f.z, f.w);
    ((uint2*)dst)[i] = o;
  }
}

// ---------------- router: logits -> top8 -> renorm weights ----------------
__global__ __launch_bounds__(256) void k_router(const float* __restrict__ x,
                                                const float* __restrict__ gate,
                                                int*   __restrict__ topk_id,
                                                float* __restrict__ topk_w,
                                                int*   __restrict__ hist) {
  __shared__ __align__(16) float xs[8][HID];     // 32 KB
  __shared__ float lg[8][NEXP];
  const int tid = threadIdx.x;
  const int t0  = blockIdx.x * 8;

  const float4* xsrc = (const float4*)(x + (size_t)t0 * HID);
  float4* xdst = (float4*)(&xs[0][0]);
#pragma unroll
  for (int i = 0; i < 8; ++i) xdst[tid + i * 256] = xsrc[tid + i * 256];
  __syncthreads();

  const int e  = tid & 63;
  const int tg = tid >> 6;                       // 0..3
  float acc0 = 0.f, acc1 = 0.f;
  const float4* gr  = (const float4*)(gate + (size_t)e * HID);
  const float4* xr0 = (const float4*)(&xs[tg][0]);
  const float4* xr1 = (const float4*)(&xs[tg + 4][0]);
  for (int j = 0; j < HID / 4; ++j) {
    float4 g = gr[j];
    float4 a = xr0[j];
    float4 b = xr1[j];
    acc0 += g.x * a.x + g.y * a.y + g.z * a.z + g.w * a.w;
    acc1 += g.x * b.x + g.y * b.y + g.z * b.z + g.w * b.w;
  }
  lg[tg][e] = acc0;
  lg[tg + 4][e] = acc1;
  __syncthreads();

  if (tid < 8) {
    const int t = t0 + tid;
    float v[NEXP];
#pragma unroll
    for (int i = 0; i < NEXP; ++i) v[i] = lg[tid][i];
    int ids[TOPK]; float lv[TOPK];
#pragma unroll
    for (int k = 0; k < TOPK; ++k) {
      int bi = 0; float bv = v[0];
      for (int i = 1; i < NEXP; ++i) if (v[i] > bv) { bv = v[i]; bi = i; }
      ids[k] = bi; lv[k] = bv; v[bi] = -3.4e38f;
    }
    // full-softmax denominator cancels after top-k renormalization
    float m = lv[0], s = 0.f, wv[TOPK];
#pragma unroll
    for (int k = 0; k < TOPK; ++k) { wv[k] = __expf(lv[k] - m); s += wv[k]; }
    float inv = 1.f / s;
#pragma unroll
    for (int k = 0; k < TOPK; ++k) {
      topk_id[t * TOPK + k] = ids[k];
      topk_w [t * TOPK + k] = wv[k] * inv;
      atomicAdd(&hist[ids[k]], 1);
    }
  }
}

// ---------------- scan + tile list ----------------
__global__ void k_scan(const int* __restrict__ hist, int* __restrict__ offsets,
                       int* __restrict__ tile_e, int* __restrict__ tile_r,
                       int* __restrict__ nTiles) {
  if (threadIdx.x == 0 && blockIdx.x == 0) {
    int acc = 0, nt = 0;
    for (int e = 0; e < NEXP; ++e) {
      offsets[e] = acc;
      int c = hist[e];
      for (int r = 0; r < c; r += MT) { tile_e[nt] = e; tile_r[nt] = r; ++nt; }
      acc += c;
    }
    offsets[NEXP] = acc;
    *nTiles = nt;
  }
}

// ---------------- scatter: permutation + inverse ----------------
__global__ void k_scatter(const int* __restrict__ topk_id,
                          const int* __restrict__ offsets,
                          int* __restrict__ cursor,
                          int* __restrict__ row_entry,
                          int* __restrict__ inv_pos) {
  int i = blockIdx.x * blockDim.x + threadIdx.x;
  if (i >= T_TOKENS * TOPK) return;
  int e = topk_id[i];
  int pos = offsets[e] + atomicAdd(&cursor[e], 1);
  row_entry[pos] = i;
  inv_pos[i] = pos;
}

// ---------------- grouped GEMM1 fused SwiGLU ----------------
// block: 256 thr (8 waves 4x2). Tile: 128 M x 64 N(inter), gate+up together.
__global__ __launch_bounds__(256) void k_gemm1(const float* __restrict__ x,
    const float* __restrict__ gup,
    const unsigned short* __restrict__ xbf,
    const unsigned short* __restrict__ gupbf,
    const int preconv,
    const int* __restrict__ row_entry,
    const int* __restrict__ offsets, const int* __restrict__ tile_e,
    const int* __restrict__ tile_r, const int* __restrict__ nTiles,
    unsigned short* __restrict__ inter) {
  const int tile = blockIdx.y;
  if (tile >= *nTiles) return;
  const int e    = tile_e[tile];
  const int row0 = tile_r[tile];
  const int p0   = offsets[e] + row0;
  const int rowsValid = (offsets[e + 1] - offsets[e]) - row0;
  const int n0   = blockIdx.x * 64;

  __shared__ unsigned short As[2][MT * 40];
  __shared__ unsigned short Bg[2][64 * 40];
  __shared__ unsigned short Bu[2][64 * 40];

  const int tid  = threadIdx.x;
  const int lane = tid & 31;
  const int wv   = tid >> 5;
  const int wm   = wv & 3;       // 0..3 (32 rows each)
  const int wn   = wv >> 2;      // 0..1 (32 cols each)

  const int lr = tid >> 1;       // 0..127
  const int lk = (tid & 1) * 16; // 0 or 16
  const bool aValid = (lr < rowsValid);
  size_t tokOff = 0;
  if (aValid) tokOff = (size_t)(row_entry[p0 + lr] >> 3) * HID;
  const float*          aSrc  = x   + tokOff + lk;
  const unsigned short* aSrcB = xbf + tokOff + lk;
  const size_t wrow = (size_t)e * (2 * INTERMED) +
                      ((lr < 64) ? (n0 + lr) : (INTERMED + n0 + (lr - 64)));
  const float*          bSrc  = gup   + wrow * HID + lk;
  const unsigned short* bSrcB = gupbf + wrow * HID + lk;
  const int bIdx = (lr & 63) * 40 + lk;
  unsigned short* bDst0 = (lr < 64) ? &Bg[0][bIdx] : &Bu[0][bIdx];
  unsigned short* bDst1 = (lr < 64) ? &Bg[1][bIdx] : &Bu[1][bIdx];

  v8f ag[2][2], au[2][2];
  v8f vz = {0.f, 0.f, 0.f, 0.f, 0.f, 0.f, 0.f, 0.f};
#pragma unroll
  for (int i = 0; i < 2; ++i)
#pragma unroll
    for (int j = 0; j < 2; ++j) { ag[i][j] = vz; au[i][j] = vz; }

  auto loadStage = [&](int buf, int kc) {
    const int kb = kc * 32;
    unsigned short* ad = &As[buf][lr * 40 + lk];
    unsigned short* bd = buf ? bDst1 : bDst0;
    if (preconv) {
      if (aValid) {
        async_copy_b128(aSrcB + kb, ad);
        async_copy_b128(aSrcB + kb + 8, ad + 8);
      } else {
        uint4 z; z.x = z.y = z.z = z.w = 0u;
        ((uint4*)ad)[0] = z; ((uint4*)ad)[1] = z;
      }
      async_copy_b128(bSrcB + kb, bd);
      async_copy_b128(bSrcB + kb + 8, bd + 8);
    } else {
      if (aValid) {
        const float4* s = (const float4*)(aSrc + kb);
        ((uint4*)ad)[0] = pk8bf(s[0], s[1]);
        ((uint4*)ad)[1] = pk8bf(s[2], s[3]);
      } else {
        uint4 z; z.x = z.y = z.z = z.w = 0u;
        ((uint4*)ad)[0] = z; ((uint4*)ad)[1] = z;
      }
      const float4* s = (const float4*)(bSrc + kb);
      ((uint4*)bd)[0] = pk8bf(s[0], s[1]);
      ((uint4*)bd)[1] = pk8bf(s[2], s[3]);
    }
  };

  auto compute = [&](int buf) {
    FragBF a[2];
#pragma unroll
    for (int mi = 0; mi < 2; ++mi) {
      const unsigned short* ap =
          &As[buf][(wm * 32 + mi * 16 + (lane & 15)) * 40 + ((lane >> 4) * 16)];
      a[mi].q[0] = *(const uint4*)ap;
      a[mi].q[1] = *(const uint4*)(ap + 8);
    }
#pragma unroll
    for (int ni = 0; ni < 2; ++ni) {
      const int coff = (wn * 32 + ni * 16 + (lane & 15)) * 40 + ((lane >> 4) * 16);
      FragBF bg, bu;
      const unsigned short* gp = &Bg[buf][coff];
      bg.q[0] = *(const uint4*)gp; bg.q[1] = *(const uint4*)(gp + 8);
      const unsigned short* up = &Bu[buf][coff];
      bu.q[0] = *(const uint4*)up; bu.q[1] = *(const uint4*)(up + 8);
#pragma unroll
      for (int mi = 0; mi < 2; ++mi) {
        ag[mi][ni] = wmma_bf16(a[mi].v, bg.v, ag[mi][ni]);
        au[mi][ni] = wmma_bf16(a[mi].v, bu.v, au[mi][ni]);
      }
    }
  };

  loadStage(0, 0);
  wait_async0();
  __syncthreads();
  for (int kc = 0; kc < HID / 32; ++kc) {
    if (kc + 1 < HID / 32) loadStage((kc + 1) & 1, kc + 1);
    compute(kc & 1);
    wait_async0();
    __syncthreads();
  }

#pragma unroll
  for (int mi = 0; mi < 2; ++mi)
#pragma unroll
    for (int ni = 0; ni < 2; ++ni)
#pragma unroll
      for (int i = 0; i < 8; ++i) {
        int r = wm * 32 + mi * 16 + ((lane >> 4) * 8) + i;
        if (r < rowsValid) {
          int c = n0 + wn * 32 + ni * 16 + (lane & 15);
          float g = ag[mi][ni][i];
          float u = au[mi][ni][i];
          float sg = g / (1.f + __expf(-g));
          inter[(size_t)(p0 + r) * INTERMED + c] = f2bf(sg * u);
        }
      }
}

// ---------------- grouped GEMM2 ----------------
// block: 256 thr (8 waves 4x2). Tile: 128 M x 128 N. K = 768.
__global__ __launch_bounds__(256) void k_gemm2(
    const unsigned short* __restrict__ inter, const float* __restrict__ down,
    const unsigned short* __restrict__ downbf, const int preconv,
    const int* __restrict__ offsets, const int* __restrict__ tile_e,
    const int* __restrict__ tile_r, const int* __restrict__ nTiles,
    float* __restrict__ second) {
  const int tile = blockIdx.y;
  if (tile >= *nTiles) return;
  const int e    = tile_e[tile];
  const int row0 = tile_r[tile];
  const int p0   = offsets[e] + row0;
  const int rowsValid = (offsets[e + 1] - offsets[e]) - row0;
  const int n0   = blockIdx.x * 128;

  __shared__ unsigned short As[2][MT * 40];
  __shared__ unsigned short Bs[2][MT * 40];

  const int tid  = threadIdx.x;
  const int lane = tid & 31;
  const int wv   = tid >> 5;
  const int wm   = wv & 3;       // 32 rows each
  const int wn   = wv >> 2;      // 0..1 (64 cols each)

  const int lr = tid >> 1;
  const int lk = (tid & 1) * 16;
  const bool aValid = (lr < rowsValid);
  const unsigned short* aSrc =
      aValid ? (inter + (size_t)(p0 + lr) * INTERMED + lk) : nullptr;
  const size_t brow = (size_t)e * HID + n0 + lr;
  const float*          bSrc  = down   + brow * INTERMED + lk;
  const unsigned short* bSrcB = downbf + brow * INTERMED + lk;

  v8f acc[2][4];
  v8f vz = {0.f, 0.f, 0.f, 0.f, 0.f, 0.f, 0.f, 0.f};
#pragma unroll
  for (int i = 0; i < 2; ++i)
#pragma unroll
    for (int j = 0; j < 4; ++j) acc[i][j] = vz;

  auto loadStage = [&](int buf, int kc) {
    const int kb = kc * 32;
    unsigned short* ad = &As[buf][lr * 40 + lk];
    if (aValid) {
      async_copy_b128(aSrc + kb, ad);
      async_copy_b128(aSrc + kb + 8, ad + 8);
    } else {
      uint4 z; z.x = z.y = z.z = z.w = 0u;
      ((uint4*)ad)[0] = z; ((uint4*)ad)[1] = z;
    }
    unsigned short* bd = &Bs[buf][lr * 40 + lk];
    if (preconv) {
      async_copy_b128(bSrcB + kb, bd);
      async_copy_b128(bSrcB + kb + 8, bd + 8);
    } else {
      const float4* s = (const float4*)(bSrc + kb);
      ((uint4*)bd)[0] = pk8bf(s[0], s[1]);
      ((uint4*)bd)[1] = pk8bf(s[2], s[3]);
    }
  };

  auto compute = [&](int buf) {
    FragBF a[2];
#pragma unroll
    for (int mi = 0; mi < 2; ++mi) {
      const unsigned short* ap =
          &As[buf][(wm * 32 + mi * 16 + (lane & 15)) * 40 + ((lane >> 4) * 16)];
      a[mi].q[0] = *(const uint4*)ap;
      a[mi].q[1] = *(const uint4*)(ap + 8);
    }
#pragma unroll
    for (int ni = 0; ni < 4; ++ni) {
      const int coff = (wn * 64 + ni * 16 + (lane & 15)) * 40 + ((lane >> 4) * 16);
      FragBF b;
      const unsigned short* bp = &Bs[buf][coff];
      b.q[0] = *(const uint4*)bp; b.q[1] = *(const uint4*)(bp + 8);
#pragma unroll
      for (int mi = 0; mi < 2; ++mi)
        acc[mi][ni] = wmma_bf16(a[mi].v, b.v, acc[mi][ni]);
    }
  };

  loadStage(0, 0);
  wait_async0();
  __syncthreads();
  for (int kc = 0; kc < INTERMED / 32; ++kc) {
    if (kc + 1 < INTERMED / 32) loadStage((kc + 1) & 1, kc + 1);
    compute(kc & 1);
    wait_async0();           // drain async writes into next buffer
    __syncthreads();
  }

#pragma unroll
  for (int mi = 0; mi < 2; ++mi)
#pragma unroll
    for (int ni = 0; ni < 4; ++ni)
#pragma unroll
      for (int i = 0; i < 8; ++i) {
        int r = wm * 32 + mi * 16 + ((lane >> 4) * 8) + i;
        if (r < rowsValid) {
          int c = n0 + wn * 64 + ni * 16 + (lane & 15);
          second[(size_t)(p0 + r) * HID + c] = acc[mi][ni][i];
        }
      }
}

// ---------------- combine ----------------
__global__ __launch_bounds__(256) void k_combine(const float* __restrict__ topk_w,
    const int* __restrict__ inv_pos, const float* __restrict__ second,
    float* __restrict__ out) {
  const int t = blockIdx.x;
  const int h = blockIdx.y * 256 + threadIdx.x;
  float acc = 0.f;
#pragma unroll
  for (int k = 0; k < TOPK; ++k) {
    float w = topk_w[t * TOPK + k];
    int   p = inv_pos[t * TOPK + k];
    acc += w * second[(size_t)p * HID + h];
  }
  out[(size_t)t * HID + h] = acc;
}

// ---------------- launch ----------------
extern "C" void kernel_launch(void* const* d_in, const int* in_sizes, int n_in,
                              void* d_out, int out_size, void* d_ws, size_t ws_size,
                              hipStream_t stream) {
  const float* x    = (const float*)d_in[0];   // [4096,1024]
  const float* gate = (const float*)d_in[1];   // [64,1024]
  const float* gup  = (const float*)d_in[2];   // [64,1536,1024]
  const float* down = (const float*)d_in[3];   // [64,1024,768]
  float* out = (float*)d_out;

  char* w = (char*)d_ws;
  int*   hist      = (int*)  (w + 0);
  int*   offsets   = (int*)  (w + 1024);
  int*   cursor    = (int*)  (w + 2048);
  int*   nTiles    = (int*)  (w + 3072);
  int*   tile_e    = (int*)  (w + 4096);
  int*   tile_r    = (int*)  (w + 8192);
  int*   topk_id   = (int*)  (w + 16384);                  // 131072 B
  float* topk_w    = (float*)(w + 147456);                 // 131072 B
  int*   row_entry = (int*)  (w + 278528);                 // 131072 B
  int*   inv_pos   = (int*)  (w + 409600);                 // 131072 B
  unsigned short* inter = (unsigned short*)(w + 540672);   // 32768*768*2
  float* second    = (float*)(w + 50872320);               // 32768*1024*4
  unsigned short* xbf    = (unsigned short*)(w + 185090048); // 4096*1024*2
  unsigned short* gupbf  = (unsigned short*)(w + 193478656); // 64*1536*1024*2
  unsigned short* downbf = (unsigned short*)(w + 394805248); // 64*1024*768*2
  const size_t NEED_BIG = 495468544;                       // end of downbf
  const int preconv = (ws_size >= NEED_BIG) ? 1 : 0;

  k_init<<<1, 64, 0, stream>>>(hist, cursor);
  k_router<<<T_TOKENS / 8, 256, 0, stream>>>(x, gate, topk_id, topk_w, hist);
  k_scan<<<1, 1, 0, stream>>>(hist, offsets, tile_e, tile_r, nTiles);
  k_scatter<<<(T_TOKENS * TOPK) / 256, 256, 0, stream>>>(topk_id, offsets, cursor,
                                                         row_entry, inv_pos);
  if (preconv) {
    k_cvt<<<4096, 256, 0, stream>>>(x,    xbf,    (long)T_TOKENS * HID / 4);
    k_cvt<<<8192, 256, 0, stream>>>(gup,  gupbf,  (long)NEXP * 2 * INTERMED * HID / 4);
    k_cvt<<<8192, 256, 0, stream>>>(down, downbf, (long)NEXP * HID * INTERMED / 4);
  }
  k_gemm1<<<dim3(INTERMED / 64, MAXTILES), 256, 0, stream>>>(
      x, gup, xbf, gupbf, preconv, row_entry, offsets, tile_e, tile_r, nTiles, inter);
  k_gemm2<<<dim3(HID / 128, MAXTILES), 256, 0, stream>>>(
      inter, down, downbf, preconv, offsets, tile_e, tile_r, nTiles, second);
  k_combine<<<dim3(T_TOKENS, HID / 256), 256, 0, stream>>>(topk_w, inv_pos, second, out);
}
